// FastDiffusionThoughtModel_32530082300191
// MI455X (gfx1250) — matrix operand order repaired and checked by reading, not compiled, not run on hardware
//
#include <hip/hip_runtime.h>
#include <math.h>

// Problem constants (match reference)
#define BB   2
#define SS   1024
#define DD   768
#define HH   12
#define LL   4
#define VV   50257
#define DFFN 3072
#define HDIM 64

// gfx1250 async-copy-to-LDS path (verified to assemble in rounds 2/3).
#ifndef USE_ASYNC_COPY
#define USE_ASYNC_COPY 1
#endif

#if USE_ASYNC_COPY
#define ASYNC_B128(ldsoff, gaddr) \
    asm volatile("global_load_async_to_lds_b128 %0, %1, off" :: "v"(ldsoff), "v"(gaddr) : "memory")
#define ASYNC_B128_O16(ldsoff, gaddr) \
    asm volatile("global_load_async_to_lds_b128 %0, %1, off offset:16" :: "v"(ldsoff), "v"(gaddr) : "memory")
#define WAIT_ASYNC() asm volatile("s_wait_asynccnt 0" ::: "memory")
#else
#define WAIT_ASYNC()
#endif

typedef __attribute__((ext_vector_type(16))) __bf16 v16bf;
typedef __attribute__((ext_vector_type(8)))  float  v8f;

union Frag { v16bf v; unsigned u[8]; };

// Hardware f32->bf16 (RNE); pairs lower to v_cvt_pk_bf16_f32.
__device__ __forceinline__ unsigned short f2bf(float f) {
    return __builtin_bit_cast(unsigned short, (__bf16)f);
}
__device__ __forceinline__ unsigned pack2(float a, float b) {
    unsigned short lo = __builtin_bit_cast(unsigned short, (__bf16)a);
    unsigned short hi = __builtin_bit_cast(unsigned short, (__bf16)b);
    return (unsigned)lo | ((unsigned)hi << 16);
}
__device__ __forceinline__ float gelu_exact(float x) {
    return 0.5f * x * (1.0f + erff(x * 0.70710678118654752f));
}
// Generic pointer to LDS: low 32 bits of the flat address are the LDS offset.
__device__ __forceinline__ unsigned ldsOff(const void* p) {
    return (unsigned)(unsigned long long)p;
}

// ---------------------------------------------------------------------------
// bf16-WMMA GEMM: C[M,N] = act(A[M,K] @ W[K,N] + bias) (+ Res if HASRES)
// block = 256 threads = 8 waves; tile 64(M) x 128(N); K-step 32;
// double-buffered LDS; A tile async-copied raw f32; W tile converted +
// transposed to bf16 [n][k] with packed b32 stores. NCHK guards odd N (vocab).
// ---------------------------------------------------------------------------
template<int ACT, bool NCHK, bool HASRES>
__global__ __launch_bounds__(256)
void gemm_bf16(const float* __restrict__ A, int lda,
               const float* __restrict__ W, int ldw,
               const float* __restrict__ bias,
               const float* __restrict__ Res,
               float* __restrict__ C,
               int M, int N, int K)
{
    __shared__ float          sA0[64 * 32],  sA1[64 * 32];     // f32 A tiles
    __shared__ unsigned short sW0[128 * 32], sW1[128 * 32];    // bf16 W^T tiles
    const int t = threadIdx.x;
    const int lane = t & 31, w = t >> 5;
    const int ml = lane & 15, hi = lane >> 4;
    const int mblk = blockIdx.y * 64;
    const int nblk = blockIdx.x * 128;
    const int m0w = (w >> 1) * 16;
    const int n0w = (w & 1) * 64;

    v8f acc[4] = {};

    auto stageTile = [&](int k0, float* __restrict__ sA, unsigned short* __restrict__ sW) {
        // ---- A: 64x32 f32 raw copy (async-to-LDS) ----
        {
            int idx = t * 8;
            int row = idx >> 5, kk = idx & 31;
            const float* ap = A + (size_t)(mblk + row) * lda + k0 + kk;
#if USE_ASYNC_COPY
            unsigned l0 = ldsOff(sA + row * 32 + kk);
            ASYNC_B128(l0, ap);
            ASYNC_B128_O16(l0, ap);
#else
            *(float4*)(sA + row * 32 + kk)     = *(const float4*)ap;
            *(float4*)(sA + row * 32 + kk + 4) = *(const float4*)(ap + 4);
#endif
        }
        // ---- W: 32x128 f32 -> bf16 transposed [n][k]; 2x2 micro-blocks so
        //      pairs pack along k into single b32 stores ----
        #pragma unroll
        for (int i = 0; i < 4; ++i) {
            int kk = (((t >> 6) + i * 4) << 1);      // 0,2,...,30
            int n  = (t & 63) << 1;                  // 0,2,...,126
            int gn = nblk + n;
            const float* wp = W + (size_t)(k0 + kk) * ldw + gn;
            float w00, w10, w01, w11;
            if constexpr (NCHK) {
                w00 = w10 = w01 = w11 = 0.0f;
                if (gn < N)     { w00 = wp[0]; w10 = wp[ldw]; }
                if (gn + 1 < N) { w01 = wp[1]; w11 = wp[ldw + 1]; }
            } else {
                float2 r0 = *(const float2*)wp;          // ldw even, gn even
                float2 r1 = *(const float2*)(wp + ldw);
                w00 = r0.x; w01 = r0.y; w10 = r1.x; w11 = r1.y;
            }
            *(unsigned*)(sW + n * 32 + kk)       = pack2(w00, w10);
            *(unsigned*)(sW + (n + 1) * 32 + kk) = pack2(w01, w11);
        }
        if (k0 + 64 < K)   // L2 prefetch two tiles ahead
            __builtin_prefetch(W + (size_t)(k0 + 64 + (t & 31)) * ldw + nblk, 0, 0);
    };

    auto computeTile = [&](const float* __restrict__ sA, const unsigned short* __restrict__ sW) {
        Frag a;
        const float* abase = sA + (m0w + ml) * 32;
        #pragma unroll
        for (int v2 = 0; v2 < 8; ++v2) {
            int kk = (hi << 4) + (v2 << 1);
            float2 x = *(const float2*)(abase + kk);
            a.u[v2] = pack2(x.x, x.y);
        }
        #pragma unroll
        for (int i = 0; i < 4; ++i) {
            Frag bf;
            const unsigned short* bbase = sW + (n0w + i * 16 + ml) * 32 + (hi << 4);
            #pragma unroll
            for (int v2 = 0; v2 < 8; ++v2)
                bf.u[v2] = *(const unsigned*)(bbase + (v2 << 1));
            acc[i] = __builtin_amdgcn_wmma_f32_16x16x32_bf16(
                false, a.v, false, bf.v, (short)0, acc[i], false, false);
        }
    };

    const int NT = K >> 5;                 // K % 64 == 0 here, so NT even >= 2
    stageTile(0, sA0, sW0);
    WAIT_ASYNC();
    __syncthreads();
    for (int kt = 0; kt < NT; kt += 2) {
        stageTile((kt + 1) << 5, sA1, sW1);
        computeTile(sA0, sW0);
        WAIT_ASYNC();
        __syncthreads();
        if (kt + 2 < NT) stageTile((kt + 2) << 5, sA0, sW0);
        computeTile(sA1, sW1);
        WAIT_ASYNC();
        __syncthreads();
    }

    // ---- epilogue: bias, activation, residual ----
    #pragma unroll
    for (int i = 0; i < 4; ++i) {
        int n = nblk + n0w + i * 16 + ml;
        if constexpr (NCHK) { if (n >= N) continue; }
        float bn = bias[n];
        #pragma unroll
        for (int r = 0; r < 8; ++r) {
            int m = mblk + m0w + r + 8 * hi;
            float val = acc[i][r] + bn;
            if constexpr (ACT == 1) val = gelu_exact(val);
            if constexpr (HASRES) val += Res[(size_t)m * N + n];
            C[(size_t)m * N + n] = val;
        }
    }
}

// ---------------------------------------------------------------------------
// Flash-style self-attention. grid = (S/128, B*H); block 256 = 8 waves.
// Q block (128x64 f32) async-copied to LDS once (coalesced), 1/sqrt(HD)
// folded into the Q->bf16 conversion. K and V tiles converted to bf16 at
// staging (1 conversion per element), so all per-chunk B-fragments are pure
// ds_load_b128 with zero VALU between the WMMAs.
// ---------------------------------------------------------------------------
__global__ __launch_bounds__(256)
void attn_kernel(const float* __restrict__ Q, const float* __restrict__ Kb,
                 const float* __restrict__ Vb, float* __restrict__ O)
{
    __shared__ float          lQ[128 * 64];      // [qrow][hd] f32 (async, once)
    __shared__ unsigned short lK[32 * 64];       // [key][hd] bf16
    __shared__ unsigned short lV[64 * 32];       // [hd][key] bf16 (transposed)
    __shared__ unsigned short lP[8 * 512];       // per-wave P staging
    const int t = threadIdx.x, lane = t & 31, w = t >> 5;
    const int ml = lane & 15, hi = lane >> 4;
    const int bh = blockIdx.y, b = bh / HH, h = bh % HH;
    const int s0 = blockIdx.x * 128;
    const int m0 = s0 + w * 16;
    unsigned short* lPw = &lP[w * 512];
    const float scale = 0.125f;                  // 1/sqrt(64), folded into Q

    // ---- async-copy the whole 128x64 Q block to LDS (coalesced) ----
    {
        int row = t >> 1, col = (t & 1) * 32;    // 32 floats (128B) per thread
        const float* qp = Q + (size_t)(b * SS + s0 + row) * DD + h * HDIM + col;
        float* lp = &lQ[row * 64 + col];
#if USE_ASYNC_COPY
        #pragma unroll
        for (int j = 0; j < 4; ++j) {
            unsigned l0 = ldsOff(lp + 8 * j);
            ASYNC_B128(l0, qp + 8 * j);
            ASYNC_B128_O16(l0, qp + 8 * j);
        }
#else
        #pragma unroll
        for (int j = 0; j < 8; ++j)
            *(float4*)(lp + 4 * j) = *(const float4*)(qp + 4 * j);
#endif
    }
    WAIT_ASYNC();
    __syncthreads();

    // ---- Q fragments (16x64 => two 16x32 A-frags), pre-scaled ----
    Frag aQ[2];
    {
        const float* qp = &lQ[(w * 16 + ml) * 64];
        #pragma unroll
        for (int ks = 0; ks < 2; ++ks)
            #pragma unroll
            for (int v2 = 0; v2 < 8; ++v2) {
                int kk = ks * 32 + (hi << 4) + (v2 << 1);
                float2 qv = *(const float2*)(qp + kk);
                aQ[ks].u[v2] = pack2(qv.x * scale, qv.y * scale);
            }
    }

    v8f o[4] = {};
    float mstat[8], ssum[8];
    #pragma unroll
    for (int r = 0; r < 8; ++r) { mstat[r] = -1e30f; ssum[r] = 0.0f; }

    for (int kc = 0; kc < SS; kc += 32) {
        __syncthreads();   // previous chunk fully consumed before restaging
        // ---- stage K as bf16 [key][hd]: 2x float4 -> 4 cvt_pk -> b128 ----
        {
            int key = t >> 3, hdb = (t & 7) * 8;
            const float* kp = Kb + (size_t)(b * SS + kc + key) * DD + h * HDIM + hdb;
            float4 k0v = *(const float4*)kp;
            float4 k1v = *(const float4*)(kp + 4);
            uint4 pk;
            pk.x = pack2(k0v.x, k0v.y); pk.y = pack2(k0v.z, k0v.w);
            pk.z = pack2(k1v.x, k1v.y); pk.w = pack2(k1v.z, k1v.w);
            *(uint4*)&lK[key * 64 + hdb] = pk;
        }
        // ---- stage V transposed bf16 [hd][key], packed along key ----
        #pragma unroll
        for (int i = 0; i < 4; ++i) {
            int kpv = ((t >> 6) + i * 4) << 1;       // 0,2,...,30
            int hd  = t & 63;
            const float* vp = Vb + (size_t)(b * SS + kc + kpv) * DD + h * HDIM + hd;
            *(unsigned*)&lV[hd * 32 + kpv] = pack2(vp[0], vp[DD]);
        }
        __syncthreads();

        // ---- scores S = (Q*scale) @ K^T : pure ds_load_b128 B-frags ----
        v8f s0v = {}, s1v = {};
        #pragma unroll
        for (int ks = 0; ks < 2; ++ks) {
            Frag bk0, bk1;
            const unsigned short* base0 = &lK[ml * 64 + ks * 32 + (hi << 4)];
            const unsigned short* base1 = &lK[(16 + ml) * 64 + ks * 32 + (hi << 4)];
            #pragma unroll
            for (int v2 = 0; v2 < 8; ++v2) {
                bk0.u[v2] = *(const unsigned*)(base0 + (v2 << 1));
                bk1.u[v2] = *(const unsigned*)(base1 + (v2 << 1));
            }
            s0v = __builtin_amdgcn_wmma_f32_16x16x32_bf16(false, aQ[ks].v, false, bk0.v, (short)0, s0v, false, false);
            s1v = __builtin_amdgcn_wmma_f32_16x16x32_bf16(false, aQ[ks].v, false, bk1.v, (short)0, s1v, false, false);
        }

        // ---- online softmax (row lives in 16 lanes of a half-wave) ----
        #pragma unroll
        for (int r = 0; r < 8; ++r) {
            float x0 = s0v[r], x1 = s1v[r];
            float mx = fmaxf(x0, x1);
            mx = fmaxf(mx, __shfl_xor(mx, 1));
            mx = fmaxf(mx, __shfl_xor(mx, 2));
            mx = fmaxf(mx, __shfl_xor(mx, 4));
            mx = fmaxf(mx, __shfl_xor(mx, 8));
            float nm = fmaxf(mstat[r], mx);
            float alpha = __expf(mstat[r] - nm);
            float p0 = __expf(x0 - nm), p1 = __expf(x1 - nm);
            float rs = p0 + p1;
            rs += __shfl_xor(rs, 1); rs += __shfl_xor(rs, 2);
            rs += __shfl_xor(rs, 4); rs += __shfl_xor(rs, 8);
            ssum[r] = ssum[r] * alpha + rs;
            mstat[r] = nm;
            #pragma unroll
            for (int f = 0; f < 4; ++f) o[f][r] *= alpha;
            int mrow = r + 8 * hi;
            lPw[mrow * 32 + ml]      = f2bf(p0);
            lPw[mrow * 32 + 16 + ml] = f2bf(p1);
        }

        // ---- O += P @ V ----
        Frag aP;
        {
            const unsigned short* base = &lPw[ml * 32];
            #pragma unroll
            for (int v2 = 0; v2 < 8; ++v2) {
                int kk = (hi << 4) + (v2 << 1);
                aP.u[v2] = *(const unsigned*)(base + kk);
            }
        }
        #pragma unroll
        for (int f = 0; f < 4; ++f) {
            Frag bv;
            const unsigned short* base = &lV[(f * 16 + ml) * 32 + (hi << 4)];
            #pragma unroll
            for (int v2 = 0; v2 < 8; ++v2)
                bv.u[v2] = *(const unsigned*)(base + (v2 << 1));
            o[f] = __builtin_amdgcn_wmma_f32_16x16x32_bf16(false, aP.v, false, bv.v, (short)0, o[f], false, false);
        }
    }

    // ---- normalize and write ----
    #pragma unroll
    for (int f = 0; f < 4; ++f) {
        int n = f * 16 + ml;
        #pragma unroll
        for (int r = 0; r < 8; ++r) {
            int m = m0 + r + 8 * hi;
            O[(size_t)(b * SS + m) * DD + h * HDIM + n] = o[f][r] / ssum[r];
        }
    }
}

// ---------------------------------------------------------------------------
// LayerNorm over rows of length D=768 (3 elems/thread, block=256).
// ---------------------------------------------------------------------------
__global__ __launch_bounds__(256)
void ln_kernel(const float* __restrict__ X, const float* __restrict__ g,
               const float* __restrict__ bta, float* __restrict__ Y)
{
    __shared__ float red[256];
    int row = blockIdx.x, t = threadIdx.x;
    const float* xr = X + (size_t)row * DD;
    float v0 = xr[t], v1 = xr[t + 256], v2 = xr[t + 512];
    red[t] = v0 + v1 + v2; __syncthreads();
    for (int off = 128; off > 0; off >>= 1) { if (t < off) red[t] += red[t + off]; __syncthreads(); }
    float mean = red[0] / (float)DD; __syncthreads();
    float d0 = v0 - mean, d1 = v1 - mean, d2 = v2 - mean;
    red[t] = d0 * d0 + d1 * d1 + d2 * d2; __syncthreads();
    for (int off = 128; off > 0; off >>= 1) { if (t < off) red[t] += red[t + off]; __syncthreads(); }
    float rstd = rsqrtf(red[0] / (float)DD + 1e-5f);
    float* yr = Y + (size_t)row * DD;
    yr[t]       = d0 * rstd * g[t]       + bta[t];
    yr[t + 256] = d1 * rstd * g[t + 256] + bta[t + 256];
    yr[t + 512] = d2 * rstd * g[t + 512] + bta[t + 512];
}

// x = LN(x + crow[b])  (cross-attention block collapsed; exact)
__global__ __launch_bounds__(256)
void addln_kernel(float* __restrict__ X, const float* __restrict__ crow,
                  const float* __restrict__ g, const float* __restrict__ bta)
{
    __shared__ float red[256];
    int row = blockIdx.x, t = threadIdx.x;
    int b = row >> 10;                 // S = 1024
    float* xr = X + (size_t)row * DD;
    const float* cr = crow + b * DD;
    float v0 = xr[t] + cr[t], v1 = xr[t + 256] + cr[t + 256], v2 = xr[t + 512] + cr[t + 512];
    red[t] = v0 + v1 + v2; __syncthreads();
    for (int off = 128; off > 0; off >>= 1) { if (t < off) red[t] += red[t + off]; __syncthreads(); }
    float mean = red[0] / (float)DD; __syncthreads();
    float d0 = v0 - mean, d1 = v1 - mean, d2 = v2 - mean;
    red[t] = d0 * d0 + d1 * d1 + d2 * d2; __syncthreads();
    for (int off = 128; off > 0; off >>= 1) { if (t < off) red[t] += red[t + off]; __syncthreads(); }
    float rstd = rsqrtf(red[0] / (float)DD + 1e-5f);
    xr[t]       = d0 * rstd * g[t]       + bta[t];
    xr[t + 256] = d1 * rstd * g[t + 256] + bta[t + 256];
    xr[t + 512] = d2 * rstd * g[t + 512] + bta[t + 512];
}

// Streaming online-logsumexp entropy confidence per row of V logits.
__global__ __launch_bounds__(256)
void entropy_kernel(const float* __restrict__ logits, float* __restrict__ ec)
{
    __shared__ float rm[256], rs[256], rt[256];
    int row = blockIdx.x, t = threadIdx.x;
    const float* lp = logits + (size_t)row * VV;
    float m = -1e30f, s = 0.0f, tt = 0.0f;
    for (int i = t; i < VV; i += 256) {
        float x = lp[i];
        if (x > m) { float f = __expf(m - x); s *= f; tt *= f; m = x; }
        float e = __expf(x - m);
        s += e; tt += x * e;
    }
    rm[t] = m; rs[t] = s; rt[t] = tt; __syncthreads();
    for (int off = 128; off > 0; off >>= 1) {
        if (t < off) {
            float m1 = rm[t], m2 = rm[t + off];
            float nm = fmaxf(m1, m2);
            float f1 = __expf(m1 - nm), f2 = __expf(m2 - nm);
            rm[t] = nm;
            rs[t] = rs[t] * f1 + rs[t + off] * f2;
            rt[t] = rt[t] * f1 + rt[t + off] * f2;
        }
        __syncthreads();
    }
    if (t == 0) {
        float lse = rm[0] + logf(rs[0]);
        float H = lse - rt[0] / rs[0];
        ec[row] = 1.0f - H / logf((float)VV);
    }
}

// Tiny matvec: out[b,n] = act(sum_k in[b,k]*W[k,n] + bias[n])  (B rows only)
template<int ACT>
__global__ void small_mv(const float* __restrict__ in, const float* __restrict__ W,
                         const float* __restrict__ bias, float* __restrict__ out,
                         int Bn, int K, int N)
{
    int idx = blockIdx.x * blockDim.x + threadIdx.x;
    if (idx >= Bn * N) return;
    int b = idx / N, n = idx % N;
    const float* ip = in + (size_t)b * K;
    float acc = bias[n];
    for (int k = 0; k < K; ++k) acc += ip[k] * W[(size_t)k * N + n];
    if (ACT == 1) acc = gelu_exact(acc);
    out[idx] = acc;
}

__global__ void timeraw_kernel(const float* __restrict__ ts, float* __restrict__ out)
{
    int idx = blockIdx.x * blockDim.x + threadIdx.x;
    if (idx >= BB * DD) return;
    int b = idx / DD, d = idx % DD;
    const int half = DD / 2;
    int dd = (d < half) ? d : d - half;
    float f = __expf(-logf(10000.0f) * (float)dd / (float)half);
    float ang = ts[b] * f;
    out[idx] = (d < half) ? sinf(ang) : cosf(ang);
}

__global__ void embed_kernel(const int* __restrict__ tokens, const float* __restrict__ tok_emb,
                             const float* __restrict__ pos_emb, const float* __restrict__ te,
                             float* __restrict__ X)
{
    int idx = blockIdx.x * blockDim.x + threadIdx.x;
    if (idx >= BB * SS * DD) return;
    int b = idx / (SS * DD);
    int rem = idx % (SS * DD);
    int s = rem / DD, d = rem % DD;
    int tok = tokens[b * SS + s];
    X[idx] = tok_emb[(size_t)tok * DD + d] + pos_emb[(size_t)s * DD + d] + te[b * DD + d];
}

__global__ void conf_kernel(const float* __restrict__ t1, const float* __restrict__ chw2,
                            const float* __restrict__ chb2, const float* __restrict__ ec,
                            float* __restrict__ out)
{
    int row = blockIdx.x * blockDim.x + threadIdx.x;
    if (row >= BB * SS) return;
    const float* tp = t1 + (size_t)row * (DD / 2);
    float acc = chb2[0];
    for (int k = 0; k < DD / 2; ++k) acc += tp[k] * chw2[k];
    float lc = 1.0f / (1.0f + __expf(-acc));
    out[row] = 0.7f * ec[row] + 0.3f * lc;
}

// ---------------------------------------------------------------------------
extern "C" void kernel_launch(void* const* d_in, const int* in_sizes, int n_in,
                              void* d_out, int out_size, void* d_ws, size_t ws_size,
                              hipStream_t stream)
{
    (void)in_sizes; (void)n_in; (void)out_size; (void)ws_size;
    const int*   tokens    = (const int*)d_in[0];
    const float* timesteps = (const float*)d_in[1];
    const float* thought   = (const float*)d_in[2];
    const float* tok_emb   = (const float*)d_in[3];
    const float* pos_emb   = (const float*)d_in[4];
    const float* tm_w1 = (const float*)d_in[5];  const float* tm_b1 = (const float*)d_in[6];
    const float* tm_w2 = (const float*)d_in[7];  const float* tm_b2 = (const float*)d_in[8];
    const float* qw = (const float*)d_in[9];   const float* qb = (const float*)d_in[10];
    const float* kw = (const float*)d_in[11];  const float* kb = (const float*)d_in[12];
    const float* vw = (const float*)d_in[13];  const float* vb = (const float*)d_in[14];
    const float* ow = (const float*)d_in[15];  const float* ob = (const float*)d_in[16];
    // d_in[17..20] (cqw,cqb,ckw,ckb) provably do not affect the output.
    const float* cvw = (const float*)d_in[21]; const float* cvb = (const float*)d_in[22];
    const float* cow = (const float*)d_in[23]; const float* cob = (const float*)d_in[24];
    const float* ln1w = (const float*)d_in[25]; const float* ln1b = (const float*)d_in[26];
    const float* ln2w = (const float*)d_in[27]; const float* ln2b = (const float*)d_in[28];
    const float* ln3w = (const float*)d_in[29]; const float* ln3b = (const float*)d_in[30];
    const float* fw1 = (const float*)d_in[31]; const float* fb1 = (const float*)d_in[32];
    const float* fw2 = (const float*)d_in[33]; const float* fb2 = (const float*)d_in[34];
    const float* onw = (const float*)d_in[35]; const float* onb = (const float*)d_in[36];
    const float* opw = (const float*)d_in[37]; const float* opb = (const float*)d_in[38];
    const float* chw1 = (const float*)d_in[39]; const float* chb1 = (const float*)d_in[40];
    const float* chw2 = (const float*)d_in[41]; const float* chb2 = (const float*)d_in[42];

    const int M = BB * SS;                 // 2048
    float* X  = (float*)d_ws;
    float* T0 = X  + (size_t)M * DD;
    float* T1 = T0 + (size_t)M * DD;
    float* T2 = T1 + (size_t)M * DD;
    float* T3 = T2 + (size_t)M * DD;
    float* R0 = T3 + (size_t)M * DD;
    float* FFb = T0;                       // DFF == 4*D, exact alias of T0..T3
    float* SM = R0 + (size_t)M * DD;
    float* te_raw = SM;
    float* tdff   = te_raw + BB * DD;
    float* te     = tdff + BB * DFFN;
    float* cv     = te + BB * DD;
    float* crow   = cv + BB * DD;
    float* ec     = crow + BB * DD;

    float* logits = (float*)d_out;
    float* conf   = logits + (size_t)M * VV;

    dim3 blk(256);
    timeraw_kernel<<<(BB * DD + 255) / 256, blk, 0, stream>>>(timesteps, te_raw);
    small_mv<1><<<(BB * DFFN + 255) / 256, blk, 0, stream>>>(te_raw, tm_w1, tm_b1, tdff, BB, DD, DFFN);
    small_mv<0><<<(BB * DD + 255) / 256, blk, 0, stream>>>(tdff, tm_w2, tm_b2, te, BB, DFFN, DD);
    embed_kernel<<<(M * DD + 255) / 256, blk, 0, stream>>>(tokens, tok_emb, pos_emb, te, X);

    dim3 gDD(DD / 128, M / 64);
    for (int i = 0; i < LL; ++i) {
        const size_t wo = (size_t)i * DD * DD, bo = (size_t)i * DD;
        gemm_bf16<0, false, false><<<gDD, blk, 0, stream>>>(X, DD, qw + wo, DD, qb + bo, nullptr, T0, M, DD, DD);
        gemm_bf16<0, false, false><<<gDD, blk, 0, stream>>>(X, DD, kw + wo, DD, kb + bo, nullptr, T1, M, DD, DD);
        gemm_bf16<0, false, false><<<gDD, blk, 0, stream>>>(X, DD, vw + wo, DD, vb + bo, nullptr, T2, M, DD, DD);
        attn_kernel<<<dim3(SS / 128, BB * HH), blk, 0, stream>>>(T0, T1, T2, T3);
        gemm_bf16<0, false, true><<<gDD, blk, 0, stream>>>(T3, DD, ow + wo, DD, ob + bo, X, R0, M, DD, DD);
        ln_kernel<<<M, blk, 0, stream>>>(R0, ln1w + bo, ln1b + bo, X);
        // cross-attention: softmax over identical keys == uniform -> output is cv
        small_mv<0><<<(BB * DD + 255) / 256, blk, 0, stream>>>(thought, cvw + wo, cvb + bo, cv, BB, DD, DD);
        small_mv<0><<<(BB * DD + 255) / 256, blk, 0, stream>>>(cv, cow + wo, cob + bo, crow, BB, DD, DD);
        addln_kernel<<<M, blk, 0, stream>>>(X, crow, ln2w + bo, ln2b + bo);
        // FFN
        gemm_bf16<1, false, false><<<dim3(DFFN / 128, M / 64), blk, 0, stream>>>(
            X, DD, fw1 + (size_t)i * DD * DFFN, DFFN, fb1 + (size_t)i * DFFN, nullptr, FFb, M, DFFN, DD);
        gemm_bf16<0, false, true><<<gDD, blk, 0, stream>>>(
            FFb, DFFN, fw2 + (size_t)i * DFFN * DD, DD, fb2 + bo, X, R0, M, DD, DFFN);
        ln_kernel<<<M, blk, 0, stream>>>(R0, ln3w + bo, ln3b + bo, X);
    }

    ln_kernel<<<M, blk, 0, stream>>>(X, onw, onb, R0);          // h -> R0
    gemm_bf16<0, true, false><<<dim3((VV + 127) / 128, M / 64), blk, 0, stream>>>(
        R0, DD, opw, VV, opb, nullptr, logits, M, VV, DD);      // 158 GF vocab GEMM
    gemm_bf16<1, false, false><<<dim3((DD / 2) / 128, M / 64), blk, 0, stream>>>(
        R0, DD, chw1, DD / 2, chb1, nullptr, T0, M, DD / 2, DD);
    entropy_kernel<<<M, blk, 0, stream>>>(logits, ec);
    conf_kernel<<<(M + 255) / 256, blk, 0, stream>>>(T0, chw2, chb2, ec, conf);
}
